// TimeAttention_topical_60988535603542
// MI455X (gfx1250) — matrix operand order repaired and checked by reading, not compiled, not run on hardware
//
#include <hip/hip_runtime.h>
#include <hip/hip_bf16.h>
#include <math.h>

// ---------------------------------------------------------------------------
// TimeAttention_topical for MI455X (gfx1250, wave32, WMMA)
// B=8 EN=512 DE=64 E=U=512 T=100
// ---------------------------------------------------------------------------

typedef __attribute__((ext_vector_type(16))) __bf16 v16bf;
typedef __attribute__((ext_vector_type(8)))  __bf16 v8bf;
typedef __attribute__((ext_vector_type(8)))  float  v8f;

#if defined(__HIP_DEVICE_COMPILE__)
# if __has_builtin(__builtin_amdgcn_tanhf)
#  define FAST_TANH(x) __builtin_amdgcn_tanhf(x)
# elif __has_builtin(__builtin_amdgcn_tanh_f32)
#  define FAST_TANH(x) __builtin_amdgcn_tanh_f32(x)
# else
#  define FAST_TANH(x) tanhf(x)
# endif
#else
# define FAST_TANH(x) tanhf(x)
#endif

#define Bb  8
#define ENN 512
#define DEE 64
#define EE  512
#define UU  512
#define TT  100
#define JT  2      // decoder positions per k_mu block (att_en reuse factor)

// ---- f32 -> bf16 flat convert ---------------------------------------------
__global__ __launch_bounds__(256) void k_f32_to_bf16(const float* __restrict__ in,
                                                     __bf16* __restrict__ out, int n) {
    int i = blockIdx.x * blockDim.x + threadIdx.x;
    if (i < n) out[i] = (__bf16)in[i];
}

// ---- f32 [R,C] -> bf16 [C,R] transpose-convert ----------------------------
__global__ __launch_bounds__(256) void k_transpose_bf16(const float* __restrict__ in,
                                                        __bf16* __restrict__ out,
                                                        int R, int C) {
    int i = blockIdx.x * blockDim.x + threadIdx.x;     // index into output [C,R]
    if (i < R * C) {
        int c = i / R;          // output row  == input column
        int r = i % R;          // output col  == input row
        out[i] = (__bf16)in[r * C + c];
    }
}

// ---- bf16 WMMA GEMM: C[M,N] = A[M,K] * B[K,N], B given transposed [N,K] ---
// one wave32 per 16x16 tile, K-loop in steps of 32 via v_wmma_f32_16x16x32_bf16
__global__ __launch_bounds__(256) void k_gemm_bf16_wmma(const __bf16* __restrict__ A,
                                                        const __bf16* __restrict__ BT,
                                                        float* __restrict__ C,
                                                        int M, int N, int K) {
    const int lane = threadIdx.x & 31;
    const int wave = threadIdx.x >> 5;
    const int tile = blockIdx.x * (blockDim.x >> 5) + wave;
    const int tilesN = N >> 4;
    const int tm = tile / tilesN;
    const int tn = tile - tm * tilesN;
    const int m0 = tm << 4, n0 = tn << 4;
    const int half = lane >> 4;     // lane half (0/1) selects K sub-block
    const int idx  = lane & 15;     // row of A / row of BT

    v8f acc = {};
    // A fragment: lane holds row (m0+idx); elems 0..7 = K(half*8 + 0..7),
    //             elems 8..15 = K(16 + half*8 + 0..7)     (ISA 16-bit A 16x32)
    const __bf16* arow = A  + (size_t)(m0 + idx) * K + half * 8;
    // B fragment (32x16, K-major per lane via BT): lane holds col (n0+idx);
    //             elems 0..7 = K(half*8 + 0..7), elems 8..15 = K(16 + half*8 + 0..7)
    const __bf16* brow = BT + (size_t)(n0 + idx) * K + half * 8;

    for (int k0 = 0; k0 < K; k0 += 32) {
        v8bf alo = *(const v8bf*)(arow + k0);
        v8bf ahi = *(const v8bf*)(arow + k0 + 16);
        v16bf a = __builtin_shufflevector(alo, ahi,
                    0,1,2,3,4,5,6,7,8,9,10,11,12,13,14,15);
        v8bf blo = *(const v8bf*)(brow + k0);
        v8bf bhi = *(const v8bf*)(brow + k0 + 16);
        v16bf b = __builtin_shufflevector(blo, bhi,
                    0,1,2,3,4,5,6,7,8,9,10,11,12,13,14,15);
        acc = __builtin_amdgcn_wmma_f32_16x16x32_bf16(
                  false, a, false, b, (short)0, acc, false, false);
    }
    // C/D layout: VGPR r -> M = r + half*8, N = n0 + idx
    float* crow = C + (size_t)(m0 + half * 8) * N + n0 + idx;
#pragma unroll
    for (int r = 0; r < 8; ++r) crow[(size_t)r * N] = acc[r];
}

// ---- topics_w[b,u] = sum_t topics[b,t] * wt[u,t] --------------------------
__global__ __launch_bounds__(256) void k_topics(const float* __restrict__ topics,
                                                const float* __restrict__ wt,
                                                float* __restrict__ tw) {
    int i = blockIdx.x * blockDim.x + threadIdx.x;   // b*U + u
    int b = i >> 9, u = i & 511;
    float acc = 0.f;
#pragma unroll 4
    for (int t = 0; t < TT; ++t)
        acc = fmaf(topics[b * TT + t], wt[u * TT + t], acc);
    tw[i] = acc;
}

// ---- mu[b,j,i] = sum_u nu[u]*tanh(att_en[b,i,u]+att_de[b,j,u]+tw[b,u]) ----
// Hot loop (134M tanh). One block per (b, j-pair, i-half). Each lane owns 16
// consecutive u values; nu[] and both j-biases live in REGISTERS (zero LDS in
// the loop). att_en streamed with b128 loads, reused for JT=2 decoder rows.
// Wave32 shfl_xor reduction (u->lane mapping is irrelevant to the sum).
__global__ __launch_bounds__(256) void k_mu(const float* __restrict__ att_en,
                                            const float* __restrict__ att_de,
                                            const float* __restrict__ tw,
                                            const float* __restrict__ nu,
                                            float* __restrict__ mu) {
    const int blk   = blockIdx.x;
    const int ihalf = blk & 1;                 // which half of the i-range
    const int bjp   = blk >> 1;                // b*32 + jpair
    const int b     = bjp >> 5;
    const int j0    = (bjp & 31) * JT;
    const int tid   = threadIdx.x;
    const int wave  = tid >> 5, lane = tid & 31;
    const int u0    = lane * 16;               // 16 consecutive u per lane

    float rnu[16], rb0[16], rb1[16];
    {
        const float4* nup = (const float4*)(nu + u0);
        const float4* twp = (const float4*)(tw + (size_t)b * UU + u0);
        const float4* d0  = (const float4*)(att_de + (size_t)(b * DEE + j0)     * UU + u0);
        const float4* d1  = (const float4*)(att_de + (size_t)(b * DEE + j0 + 1) * UU + u0);
#pragma unroll
        for (int q = 0; q < 4; ++q) {
            float4 n4 = nup[q], t4 = twp[q], a4 = d0[q], b4 = d1[q];
            rnu[q*4+0] = n4.x; rnu[q*4+1] = n4.y; rnu[q*4+2] = n4.z; rnu[q*4+3] = n4.w;
            rb0[q*4+0] = a4.x + t4.x; rb0[q*4+1] = a4.y + t4.y;
            rb0[q*4+2] = a4.z + t4.z; rb0[q*4+3] = a4.w + t4.w;
            rb1[q*4+0] = b4.x + t4.x; rb1[q*4+1] = b4.y + t4.y;
            rb1[q*4+2] = b4.z + t4.z; rb1[q*4+3] = b4.w + t4.w;
        }
    }

    float* mu0 = mu + (size_t)(b * DEE + j0) * ENN;
    float* mu1 = mu0 + ENN;
    const float* ae = att_en + (size_t)b * ENN * UU;

    const int iend = ihalf * 256 + 256;
    for (int i = ihalf * 256 + wave; i < iend; i += 8) {
        const float4* r = (const float4*)(ae + (size_t)i * UU + u0);
        float v[16];
#pragma unroll
        for (int q = 0; q < 4; ++q) {
            float4 t = r[q];
            v[q*4+0] = t.x; v[q*4+1] = t.y; v[q*4+2] = t.z; v[q*4+3] = t.w;
        }
        float acc0 = 0.f, acc1 = 0.f;
#pragma unroll
        for (int u = 0; u < 16; ++u) {
            acc0 = fmaf(FAST_TANH(v[u] + rb0[u]), rnu[u], acc0);
            acc1 = fmaf(FAST_TANH(v[u] + rb1[u]), rnu[u], acc1);
        }
#pragma unroll
        for (int off = 16; off > 0; off >>= 1) {
            acc0 += __shfl_xor(acc0, off, 32);
            acc1 += __shfl_xor(acc1, off, 32);
        }
        if (lane == 0) { mu0[i] = acc0; mu1[i] = acc1; }
    }
}

// ---- softmax over i (row of 512) + sigmoid --------------------------------
__global__ __launch_bounds__(256) void k_softmax_pgen(const float* __restrict__ mu,
                                                      float* __restrict__ alphas,
                                                      float* __restrict__ pgen) {
    __shared__ float red[256];
    const int row = blockIdx.x, tid = threadIdx.x;
    const float* m = mu + (size_t)row * ENN;
    float x0 = m[tid], x1 = m[tid + 256];

    red[tid] = fmaxf(x0, x1);
    __syncthreads();
    for (int s = 128; s > 0; s >>= 1) {
        if (tid < s) red[tid] = fmaxf(red[tid], red[tid + s]);
        __syncthreads();
    }
    float mx = red[0];
    __syncthreads();

    float e0 = __expf(x0 - mx), e1 = __expf(x1 - mx);
    red[tid] = e0 + e1;
    __syncthreads();
    for (int s = 128; s > 0; s >>= 1) {
        if (tid < s) red[tid] += red[tid + s];
        __syncthreads();
    }
    float inv = __frcp_rn(red[0]);

    alphas[(size_t)row * ENN + tid]       = e0 * inv;
    alphas[(size_t)row * ENN + tid + 256] = e1 * inv;
    pgen[(size_t)row * ENN + tid]         = 1.f / (1.f + __expf(-x0));
    pgen[(size_t)row * ENN + tid + 256]   = 1.f / (1.f + __expf(-x1));
}

// ---- output[b,j,e] = de_seq[b,j,e] + sum_i alphas[b,j,i]*en_seq[b,i,e] ----
__global__ __launch_bounds__(256) void k_output(const float* __restrict__ alphas,
                                                const float* __restrict__ en_seq,
                                                const float* __restrict__ de_seq,
                                                float* __restrict__ out) {
    __shared__ float al[ENN];
    const int bj = blockIdx.x;
    const int b  = bj >> 6;
    const int tid = threadIdx.x;
    al[tid]       = alphas[(size_t)bj * ENN + tid];
    al[tid + 256] = alphas[(size_t)bj * ENN + tid + 256];
    __syncthreads();

    const float* en = en_seq + (size_t)b * ENN * EE;
    float a0 = 0.f, a1 = 0.f;
    for (int i = 0; i < ENN; ++i) {
        float w = al[i];
        const float* r = en + (size_t)i * EE;
        a0 = fmaf(w, r[tid], a0);
        a1 = fmaf(w, r[tid + 256], a1);
    }
    out[(size_t)bj * EE + tid]       = de_seq[(size_t)bj * EE + tid]       + a0;
    out[(size_t)bj * EE + tid + 256] = de_seq[(size_t)bj * EE + tid + 256] + a1;
}

// ---------------------------------------------------------------------------
extern "C" void kernel_launch(void* const* d_in, const int* in_sizes, int n_in,
                              void* d_out, int out_size, void* d_ws, size_t ws_size,
                              hipStream_t stream) {
    const float* en_seq = (const float*)d_in[0];   // [8,512,512]
    const float* de_seq = (const float*)d_in[1];   // [8,64,512]
    const float* topics = (const float*)d_in[2];   // [8,100]
    const float* w_en   = (const float*)d_in[3];   // [512,512]
    const float* w_de   = (const float*)d_in[4];   // [512,512]
    const float* nu     = (const float*)d_in[5];   // [512]
    const float* wt     = (const float*)d_in[6];   // [512,100]

    const size_t MB = 1u << 20;
    char* ws = (char*)d_ws;
    __bf16* en_bf  = (__bf16*)(ws);                              // 4.0 MiB
    __bf16* de_bf  = (__bf16*)(ws + 4 * MB);                     // 0.5 MiB
    __bf16* wenT   = (__bf16*)(ws + 4 * MB + 512 * 1024);        // 0.5 MiB
    __bf16* wdeT   = (__bf16*)(ws + 5 * MB);                     // 0.5 MiB
    float*  att_en = (float*)(ws + 5 * MB + 512 * 1024);         // 8.0 MiB
    float*  att_de = (float*)(ws + 13 * MB + 512 * 1024);        // 1.0 MiB
    float*  tw     = (float*)(ws + 14 * MB + 512 * 1024);        // 16 KiB
    float*  mu     = (float*)(ws + 14 * MB + 512 * 1024 + 16 * 1024); // 1.0 MiB

    float* out_output = (float*)d_out;                 // [8,64,512]
    float* out_alphas = out_output + Bb * DEE * EE;    // [8,64,512]
    float* out_pgen   = out_alphas + Bb * DEE * ENN;   // [8,64,512]

    // precision-convert activations / pre-transpose weights for WMMA
    k_f32_to_bf16 <<<(Bb*ENN*EE) / 256, 256, 0, stream>>>(en_seq, en_bf, Bb*ENN*EE);
    k_f32_to_bf16 <<<(Bb*DEE*EE) / 256, 256, 0, stream>>>(de_seq, de_bf, Bb*DEE*EE);
    k_transpose_bf16<<<(EE*UU) / 256, 256, 0, stream>>>(w_en, wenT, EE, UU);
    k_transpose_bf16<<<(EE*UU) / 256, 256, 0, stream>>>(w_de, wdeT, EE, UU);

    // projections via v_wmma_f32_16x16x32_bf16 (f32 accumulate)
    // en: M=4096 N=512 -> 8192 tiles -> 1024 blocks of 8 waves
    k_gemm_bf16_wmma<<<1024, 256, 0, stream>>>(en_bf, wenT, att_en, Bb*ENN, UU, EE);
    // de: M=512  N=512 -> 1024 tiles -> 128 blocks
    k_gemm_bf16_wmma<<<128,  256, 0, stream>>>(de_bf, wdeT, att_de, Bb*DEE, UU, EE);

    k_topics<<<(Bb*UU) / 256, 256, 0, stream>>>(topics, wt, tw);

    // fused tanh + dot(nu) reduction (the 134M-tanh hot loop)
    // grid = B * (DE/JT) * 2 i-halves = 8*32*2 = 512 blocks
    k_mu<<<Bb * (DEE / JT) * 2, 256, 0, stream>>>(att_en, att_de, tw, nu, mu);

    k_softmax_pgen<<<Bb*DEE, 256, 0, stream>>>(mu, out_alphas, out_pgen);

    k_output<<<Bb*DEE, 256, 0, stream>>>(out_alphas, en_seq, de_seq, out_output);
}
